// KappaModel_52347061403802
// MI455X (gfx1250) — matrix operand order repaired and checked by reading, not compiled, not run on hardware
//
#include <hip/hip_runtime.h>
#include <hip/hip_bf16.h>
#include <cstdint>
#include <cstddef>

#define B_SZ  8
#define SEQ   2048
#define MTOK  (B_SZ * SEQ)     // 16384 tokens
#define DM    128
#define DI    256
#define DS    8
#define DTR   8
#define H1N   32

typedef __attribute__((ext_vector_type(16))) _Float16 v16h;
typedef __attribute__((ext_vector_type(8)))  _Float16 v8h;
typedef __attribute__((ext_vector_type(8)))  float    v8f;

// ---------------------------------------------------------------------------
// Register-blocked WMMA GEMM:
//   C[M x N] (f32) = A[M x K] (f16, row major) * W[N x K]^T
// One wave computes a 16 x (16*NT) strip. Per K-step: all fragment loads are
// issued FIRST (one clause of global_load_b128), then NT WMMAs run back to
// back -> the scheduler can stage s_wait_loadcnt and hide VMEM latency under
// the matrix pipe instead of draining the load queue before every WMMA.
// Fragment layouts as loaded here (wave32; A and B share one k order):
//   A 16x32 f16 : lane m = lane&15 ; halves e -> k = hi*16 + e
//   B 32x16 f16 : lane n = lane&15 ; halves e -> k = hi*16 + e
//   C 16x16 f32 : VGPR r, lane l   -> row = (l>>4)*8 + r, col = l&15
// ---------------------------------------------------------------------------
template <int KD, int NT>
__global__ __launch_bounds__(32)
void k_gemm_wmma(const _Float16* __restrict__ A,
                 const _Float16* __restrict__ W,
                 float* __restrict__ C,
                 int Ndim) {
  const int ntn  = Ndim / (16 * NT);
  const int tile = blockIdx.x;
  const int tm   = (tile / ntn) << 4;
  const int tn0  = (tile % ntn) * (16 * NT);
  const int lane = threadIdx.x;
  const int hi   = (lane >> 4) & 1;
  const int lo   = lane & 15;

  const _Float16* Arow = A + (size_t)(tm + lo) * KD;
  const _Float16* Wrow = W + (size_t)(tn0 + lo) * KD;

  v8f zero = {};
  v8f acc[NT];
#pragma unroll
  for (int j = 0; j < NT; ++j) acc[j] = zero;

#pragma unroll
  for (int k0 = 0; k0 < KD; k0 += 32) {
    // ---- issue ALL loads for this K-step first (one big VMEM clause) ----
    v8h alo = *(const v8h*)(Arow + k0 + hi * 16);
    v8h ahi = *(const v8h*)(Arow + k0 + hi * 16 + 8);
    v16h bfrag[NT];
#pragma unroll
    for (int j = 0; j < NT; ++j)
      bfrag[j] = *(const v16h*)(Wrow + (size_t)j * 16 * KD + k0 + hi * 16);

    v16h a;
#pragma unroll
    for (int i = 0; i < 8; ++i) { a[i] = alo[i]; a[8 + i] = ahi[i]; }

    // ---- then the WMMA chain, back to back ----
#pragma unroll
    for (int j = 0; j < NT; ++j)
      acc[j] = __builtin_amdgcn_wmma_f32_16x16x32_f16(
          false, a, false, bfrag[j], (short)0, acc[j], false, false);
  }

#pragma unroll
  for (int j = 0; j < NT; ++j) {
    float* Crow = C + (size_t)(tm + hi * 8) * Ndim + tn0 + j * 16 + lo;
#pragma unroll
    for (int r = 0; r < 8; ++r) Crow[(size_t)r * Ndim] = acc[j][r];
  }
}

// ---------------------------------------------------------------------------
// Elementwise / small-K kernels
// ---------------------------------------------------------------------------
__global__ void k_f32_to_f16(const float* __restrict__ s, _Float16* __restrict__ d, int n) {
  int i = blockIdx.x * blockDim.x + threadIdx.x;
  if (i < n) d[i] = (_Float16)s[i];
}

__global__ void k_zero_f16(_Float16* __restrict__ d, int n) {
  int i = blockIdx.x * blockDim.x + threadIdx.x;
  if (i < n) d[i] = (_Float16)0.0f;
}

// u = tanh(x @ proj0_w.T + proj0_b)  -> f16 (K=3, scalar)
__global__ void k_proj0(const float* __restrict__ x, const float* __restrict__ w,
                        const float* __restrict__ b, _Float16* __restrict__ u16) {
  int i = blockIdx.x * blockDim.x + threadIdx.x;        // m*128 + j
  if (i >= MTOK * DM) return;
  int m = i >> 7, j = i & 127;
  float s = b[j] + x[m * 3] * w[j * 3] + x[m * 3 + 1] * w[j * 3 + 1]
          + x[m * 3 + 2] * w[j * 3 + 2];
  u16[i] = (_Float16)tanhf(s);
}

// depthwise causal conv (k=4) on xs half of xz, + bias, SiLU -> f32 + f16
__global__ void k_conv_silu(const float* __restrict__ xz, const float* __restrict__ cw,
                            const float* __restrict__ cb,
                            float* __restrict__ xs32, _Float16* __restrict__ xs16) {
  int i = blockIdx.x * blockDim.x + threadIdx.x;        // m*256 + d
  if (i >= MTOK * DI) return;
  int m = i >> 8, d = i & 255;
  int l = m & (SEQ - 1);
  float s = cb[d];
#pragma unroll
  for (int q = 0; q < 4; ++q) {
    int lt = l - 3 + q;
    if (lt >= 0) s += cw[d * 4 + q] * xz[(size_t)(m - 3 + q) * 512 + d];
  }
  float a = s / (1.f + __expf(-s));
  xs32[i] = a;
  xs16[i] = (_Float16)a;
}

// delta = softplus(dt @ dt_proj_w.T + dt_proj_b)   (K=8, scalar)
__global__ void k_delta(const float* __restrict__ xdbl, const float* __restrict__ dtw,
                        const float* __restrict__ dtb, float* __restrict__ delta) {
  int i = blockIdx.x * blockDim.x + threadIdx.x;        // m*256 + d
  if (i >= MTOK * DI) return;
  int m = i >> 8, d = i & 255;
  const float* xr = xdbl + (size_t)m * 32;              // cols 0..7 = dt
  const float* wr = dtw + d * DTR;
  float s = dtb[d];
#pragma unroll
  for (int r = 0; r < 8; ++r) s += xr[r] * wr[r];
  delta[i] = (s > 20.f) ? s : log1pf(__expf(s));
}

// Selective scan: one wave per (batch, 4 channels); lane = (d_sub<<3) | n.
// h[d,n] lives in a register; y[d] = sum_n h*C via 8-lane shuffle reduce.
__global__ __launch_bounds__(32)
void k_scan(const float* __restrict__ delta, const float* __restrict__ xs,
            const float* __restrict__ xdbl,  const float* __restrict__ A_log,
            float* __restrict__ y) {
  int blk  = blockIdx.x;
  int b    = blk >> 6;                 // 64 blocks per batch
  int dgrp = blk & 63;
  int lane = threadIdx.x;
  int n    = lane & 7;
  int d    = dgrp * 4 + (lane >> 3);
  float negA = -__expf(A_log[d * DS + n]);

  const size_t base = (size_t)b * SEQ;
  const float* dp = delta + base * DI + d;
  const float* xp = xs    + base * DI + d;
  const float* bp = xdbl  + base * 32 + 8  + n;
  const float* cp = xdbl  + base * 32 + 16 + n;
  float*       yp = y     + base * DI + d;

  float h = 0.f;
  for (int t = 0; t < SEQ; ++t) {
    float dv = dp[(size_t)t * DI];
    float xv = xp[(size_t)t * DI];
    float Bv = bp[(size_t)t * 32];
    float Cv = cp[(size_t)t * 32];
    h = __expf(dv * negA) * h + dv * xv * Bv;
    float p = h * Cv;
    p += __shfl_xor(p, 1, 32);
    p += __shfl_xor(p, 2, 32);
    p += __shfl_xor(p, 4, 32);
    if (n == 0) yp[(size_t)t * DI] = p;
  }
}

// y2 = (y + Dp*xs) * silu(z)   -> f16 for out_proj
__global__ void k_ypost(const float* __restrict__ y, const float* __restrict__ xs32,
                        const float* __restrict__ xz, const float* __restrict__ Dp,
                        _Float16* __restrict__ y216) {
  int i = blockIdx.x * blockDim.x + threadIdx.x;        // m*256 + d
  if (i >= MTOK * DI) return;
  int m = i >> 8, d = i & 255;
  float z  = xz[(size_t)m * 512 + 256 + d];
  float sz = z / (1.f + __expf(-z));
  y216[i]  = (_Float16)((y[i] + Dp[d] * xs32[i]) * sz);
}

__global__ void k_elu_to_f16(const float* __restrict__ s, _Float16* __restrict__ d, int n) {
  int i = blockIdx.x * blockDim.x + threadIdx.x;
  if (i >= n) return;
  float v = s[i];
  d[i] = (_Float16)(v > 0.f ? v : expm1f(v));
}

__global__ void k_bias_dual(const float* __restrict__ s, const float* __restrict__ bias,
                            float* __restrict__ d32, _Float16* __restrict__ d16,
                            int rowmask, int n) {
  int i = blockIdx.x * blockDim.x + threadIdx.x;
  if (i >= n) return;
  float v = s[i] + bias[i & rowmask];
  d32[i] = v;
  d16[i] = (_Float16)v;
}

__global__ void k_bias_elu_f16(const float* __restrict__ s, const float* __restrict__ bias,
                               _Float16* __restrict__ d16, int rowmask, int n) {
  int i = blockIdx.x * blockDim.x + threadIdx.x;
  if (i >= n) return;
  float v = s[i] + bias[i & rowmask];
  d16[i] = (_Float16)(v > 0.f ? v : expm1f(v));
}

__global__ void k_bias_f32(const float* __restrict__ s, const float* __restrict__ bias,
                           float* __restrict__ d, int rowmask, int n) {
  int i = blockIdx.x * blockDim.x + threadIdx.x;
  if (i >= n) return;
  d[i] = s[i] + bias[i & rowmask];
}

// final head: proj2 over concat [x(3), mamba_out(128), out1(32)], sine warp, ratios
__global__ void k_final(const float* __restrict__ x,  const float* __restrict__ mo,
                        const float* __restrict__ o1, const float* __restrict__ W2,
                        const float* __restrict__ b2, float* __restrict__ out) {
  int m = blockIdx.x * blockDim.x + threadIdx.x;
  if (m >= MTOK) return;
  float x0 = x[m * 3], x1 = x[m * 3 + 1], x2 = x[m * 3 + 2];
  const float* mrow = mo + (size_t)m * DM;
  const float* orow = o1 + (size_t)m * H1N;
  float o[5];
#pragma unroll
  for (int j = 0; j < 5; ++j) {
    const float* wj = W2 + j * 163;
    float s = b2[j] + x0 * wj[0] + x1 * wj[1] + x2 * wj[2];
    for (int k = 0; k < DM; ++k)  s += mrow[k] * wj[3 + k];
    for (int k = 0; k < H1N; ++k) s += orow[k] * wj[131 + k];
    o[j] = s;
  }
  const float PI = 3.14159265358979323846f;
#pragma unroll
  for (int it = 0; it < 2; ++it)
#pragma unroll
    for (int j = 0; j < 5; ++j) o[j] += sinf(o[j] * PI) * 0.5f;
#pragma unroll
  for (int j = 0; j < 5; ++j) o[j] = (o[j] + 1.f) * 0.5f;
  o[2] += 1.f; o[3] += 1.f; o[4] += 1.f;
  out[m * 3 + 0] = x0 + o[0];
  out[m * 3 + 1] = x1 + x2 * (o[1] / o[2]);
  out[m * 3 + 2] = x2 * (o[3] / o[4]);
}

// ---------------------------------------------------------------------------
extern "C" void kernel_launch(void* const* d_in, const int* in_sizes, int n_in,
                              void* d_out, int out_size, void* d_ws, size_t ws_size,
                              hipStream_t stream) {
  (void)in_sizes; (void)n_in; (void)out_size; (void)ws_size;
  const float* x         = (const float*)d_in[0];
  const float* proj0_w   = (const float*)d_in[1];
  const float* proj0_b   = (const float*)d_in[2];
  const float* in_proj_w = (const float*)d_in[3];
  const float* conv_w    = (const float*)d_in[4];
  const float* conv_b    = (const float*)d_in[5];
  const float* x_proj_w  = (const float*)d_in[6];
  const float* dt_proj_w = (const float*)d_in[7];
  const float* dt_proj_b = (const float*)d_in[8];
  const float* A_log     = (const float*)d_in[9];
  const float* Dp        = (const float*)d_in[10];
  const float* out_proj_w= (const float*)d_in[11];
  const float* mlin_w    = (const float*)d_in[12];
  const float* mlin_b    = (const float*)d_in[13];
  const float* proj1_w   = (const float*)d_in[14];
  const float* proj1_b   = (const float*)d_in[15];
  const float* lin1_w    = (const float*)d_in[16];
  const float* lin1_b    = (const float*)d_in[17];
  const float* proj2_w   = (const float*)d_in[18];
  const float* proj2_b   = (const float*)d_in[19];
  float* out = (float*)d_out;

  char* ws = (char*)d_ws;
  size_t off = 0;
  auto alloc = [&](size_t bytes) -> void* {
    void* p = ws + off;
    off += (bytes + 255) & ~(size_t)255;
    return p;
  };

  _Float16* u16     = (_Float16*)alloc((size_t)MTOK * DM * 2);
  float*    xz      = (float*)   alloc((size_t)MTOK * 512 * 4);
  float*    xs32    = (float*)   alloc((size_t)MTOK * DI * 4);
  _Float16* xs16    = (_Float16*)alloc((size_t)MTOK * DI * 2);
  float*    xdbl    = (float*)   alloc((size_t)MTOK * 32 * 4);
  float*    delta   = (float*)   alloc((size_t)MTOK * DI * 4);
  float*    yscan   = (float*)   alloc((size_t)MTOK * DI * 4);
  _Float16* y216    = (_Float16*)alloc((size_t)MTOK * DI * 2);
  float*    opraw   = (float*)   alloc((size_t)MTOK * DM * 4);   // out_proj result
  _Float16* t116    = (_Float16*)alloc((size_t)MTOK * DM * 2);   // elu(out_proj)
  float*    mlraw   = (float*)   alloc((size_t)MTOK * DM * 4);   // mlin result
  float*    mo32    = (float*)   alloc((size_t)MTOK * DM * 4);   // mamba_out
  _Float16* mo16    = (_Float16*)alloc((size_t)MTOK * DM * 2);
  float*    p1raw   = (float*)   alloc((size_t)MTOK * H1N * 4);
  _Float16* h116    = (_Float16*)alloc((size_t)MTOK * H1N * 2);
  float*    l1raw   = (float*)   alloc((size_t)MTOK * H1N * 4);
  float*    out1    = (float*)   alloc((size_t)MTOK * H1N * 4);
  _Float16* w_in16  = (_Float16*)alloc((size_t)512 * DM * 2);
  _Float16* w_xp16  = (_Float16*)alloc((size_t)32 * DI * 2);     // padded 24->32 rows
  _Float16* w_op16  = (_Float16*)alloc((size_t)DM * DI * 2);
  _Float16* w_ml16  = (_Float16*)alloc((size_t)DM * DM * 2);
  _Float16* w_p116  = (_Float16*)alloc((size_t)H1N * DM * 2);
  _Float16* w_l116  = (_Float16*)alloc((size_t)H1N * H1N * 2);

  const int T = 256;
  auto g = [&](int n) { return dim3((n + T - 1) / T); };

  // --- weight conversion (f32 -> f16), x_proj padded with zero rows ---
  k_f32_to_f16<<<g(512 * DM), T, 0, stream>>>(in_proj_w, w_in16, 512 * DM);
  k_zero_f16  <<<g(32 * DI),  T, 0, stream>>>(w_xp16, 32 * DI);
  k_f32_to_f16<<<g(24 * DI),  T, 0, stream>>>(x_proj_w, w_xp16, 24 * DI);
  k_f32_to_f16<<<g(DM * DI),  T, 0, stream>>>(out_proj_w, w_op16, DM * DI);
  k_f32_to_f16<<<g(DM * DM),  T, 0, stream>>>(mlin_w, w_ml16, DM * DM);
  k_f32_to_f16<<<g(H1N * DM), T, 0, stream>>>(proj1_w, w_p116, H1N * DM);
  k_f32_to_f16<<<g(H1N * H1N),T, 0, stream>>>(lin1_w, w_l116, H1N * H1N);

  // --- 1) u = tanh(proj0(x))  (f16) ---
  k_proj0<<<g(MTOK * DM), T, 0, stream>>>(x, proj0_w, proj0_b, u16);

  // --- 2) xz = u @ in_proj_w^T   [16384 x 512], K=128  (WMMA, 16x64 strips) ---
  k_gemm_wmma<128, 4><<<dim3((MTOK / 16) * (512 / 64)), 32, 0, stream>>>(
      u16, w_in16, xz, 512);

  // --- 3) depthwise conv + SiLU -> xs ---
  k_conv_silu<<<g(MTOK * DI), T, 0, stream>>>(xz, conv_w, conv_b, xs32, xs16);

  // --- 4) x_dbl = xs @ x_proj^T   [16384 x 32(pad)], K=256  (WMMA, 16x32) ---
  k_gemm_wmma<256, 2><<<dim3((MTOK / 16) * (32 / 32)), 32, 0, stream>>>(
      xs16, w_xp16, xdbl, 32);

  // --- 5) delta = softplus(dt_proj(dt)) ---
  k_delta<<<g(MTOK * DI), T, 0, stream>>>(xdbl, dt_proj_w, dt_proj_b, delta);

  // --- 6) selective scan (512 waves, register-resident state) ---
  k_scan<<<dim3(B_SZ * 64), 32, 0, stream>>>(delta, xs32, xdbl, A_log, yscan);

  // --- 7) y2 = (y + Dp*xs)*silu(z) -> f16 ---
  k_ypost<<<g(MTOK * DI), T, 0, stream>>>(yscan, xs32, xz, Dp, y216);

  // --- 8) out_proj: [16384 x 128], K=256  (WMMA, 16x64 strips) ---
  k_gemm_wmma<256, 4><<<dim3((MTOK / 16) * (DM / 64)), 32, 0, stream>>>(
      y216, w_op16, opraw, DM);

  // --- 9) elu -> f16 ---
  k_elu_to_f16<<<g(MTOK * DM), T, 0, stream>>>(opraw, t116, MTOK * DM);

  // --- 10) mlin: [16384 x 128], K=128 (WMMA); + bias -> mamba_out f32/f16 ---
  k_gemm_wmma<128, 4><<<dim3((MTOK / 16) * (DM / 64)), 32, 0, stream>>>(
      t116, w_ml16, mlraw, DM);
  k_bias_dual<<<g(MTOK * DM), T, 0, stream>>>(mlraw, mlin_b, mo32, mo16,
                                              DM - 1, MTOK * DM);

  // --- 11) proj1: [16384 x 32], K=128 (WMMA); + bias, elu -> f16 ---
  k_gemm_wmma<128, 2><<<dim3((MTOK / 16) * (H1N / 32)), 32, 0, stream>>>(
      mo16, w_p116, p1raw, H1N);
  k_bias_elu_f16<<<g(MTOK * H1N), T, 0, stream>>>(p1raw, proj1_b, h116,
                                                  H1N - 1, MTOK * H1N);

  // --- 12) lin1: [16384 x 32], K=32 (WMMA); + bias -> out1 f32 ---
  k_gemm_wmma<32, 2><<<dim3((MTOK / 16) * (H1N / 32)), 32, 0, stream>>>(
      h116, w_l116, l1raw, H1N);
  k_bias_f32<<<g(MTOK * H1N), T, 0, stream>>>(l1raw, lin1_b, out1,
                                              H1N - 1, MTOK * H1N);

  // --- 13) proj2 over concat + sine warp + final ratios ---
  k_final<<<dim3(MTOK / 128), 128, 0, stream>>>(x, mo32, out1, proj2_w, proj2_b, out);
}
